// RFBMultiHeadAttn_V2_23991687316193
// MI455X (gfx1250) — matrix operand order repaired and checked by reading, not compiled, not run on hardware
//
#include <hip/hip_runtime.h>
#include <hip/hip_bf16.h>

typedef __bf16 bf16;
typedef __attribute__((ext_vector_type(16))) __bf16 v16bf;
typedef __attribute__((ext_vector_type(8)))  __bf16 v8bf;
typedef __attribute__((ext_vector_type(8)))  float  v8f;

#define B_SZ   1024
#define IN_DIM 2048
#define NH     16
#define D      64
#define INNER  1024
#define TF     4
#define NC     2
#define EPS    1e-5f
#define SLOPE  0.01f
#define CHUNK  256      /* B_SZ / TF rows per pooled chunk */

/* ---------------- workspace layout (byte offsets) ---------------- */
#define OFF_QBF   ((size_t)0)                 /* 1024*2048 bf16 = 4 MB  */
#define OFF_KVBF  ((size_t)4  << 20)          /* 1024*2048 bf16 = 4 MB  */
#define OFF_WQBF  ((size_t)8  << 20)          /* 1024*2048 bf16 = 4 MB  */
#define OFF_WKVBF ((size_t)12 << 20)          /* 2048*2048 bf16 = 8 MB  */
#define OFF_WVS   ((size_t)20 << 20)          /* 16*2048 bf16   = 64 KB */
#define OFF_SIM   (OFF_WVS  + ((size_t)64 << 10))   /* 1024*16 f32 */
#define OFF_VS    (OFF_SIM  + ((size_t)64 << 10))   /* 1024*16 f32 */
#define OFF_ATT   (OFF_VS   + ((size_t)64 << 10))   /* 1024*16 f32 */
#define OFF_ACH   (OFF_ATT  + ((size_t)64 << 10))   /* 4*16 f32    */
#define OFF_BKVS  (OFF_ACH  + (size_t)1024)         /* 16 f32      */
#define OFF_X     (OFF_BKVS + (size_t)1024)         /* 1024*4 f32  */

/* ---------------- fp32 -> bf16 conversion prepass ---------------- */
__global__ void cvt_bf16_kernel(const float* __restrict__ src,
                                bf16* __restrict__ dst, int n) {
  int i = blockIdx.x * blockDim.x + threadIdx.x;
  if (i < n) dst[i] = (bf16)src[i];
}

/* Wvsum[n,m] = sum_d Wkv[n*128+64+d, m]  (v-half rows, head-interleaved)
   bkvvs[n]  = sum_d bkv[n*128+64+d]                                      */
__global__ void wvsum_kernel(const float* __restrict__ Wkv,
                             bf16* __restrict__ wvs,
                             float* __restrict__ bkvvs,
                             const float* __restrict__ bkv) {
  int idx = blockIdx.x * blockDim.x + threadIdx.x;
  if (idx < NH * IN_DIM) {
    int n = idx >> 11, m = idx & (IN_DIM - 1);
    float s = 0.f;
    for (int d = 0; d < D; ++d)
      s += Wkv[(size_t)(n * 128 + 64 + d) * IN_DIM + m];
    wvs[(size_t)n * IN_DIM + m] = (bf16)s;
  }
  if (idx < NH) {
    float s = 0.f;
    for (int d = 0; d < D; ++d) s += bkv[idx * 128 + 64 + d];
    bkvvs[idx] = s;
  }
}

/* ---------------- WMMA fragment loaders (contiguous k order in A and B) -------- */
/* A (16x32 bf16): lane<16 holds row M=lane, K = 0..15 contiguous;
                   lane>=16 same row, K = 16..31.                        */
__device__ __forceinline__ v16bf load_a_frag(const bf16* rowp, int m0, int lane) {
  const int kb = (lane < 16) ? 0 : 16;
  union { v16bf v; struct { v8bf lo, hi; } s; } u;
  u.s.lo = *(const v8bf*)(rowp + m0 + kb);
  u.s.hi = *(const v8bf*)(rowp + m0 + kb + 8);
  return u.v;
}
/* B (32x16 bf16): lane<16 holds col N=lane, K=0..15 contiguous;
                   lane>=16 col N=lane-16, K=16..31. One 32B load.       */
__device__ __forceinline__ v16bf load_b_frag(const bf16* colp, int m0, int lane) {
  const int ko = (lane < 16) ? 0 : 16;
  return *(const v16bf*)(colp + m0 + ko);
}

/* ---------------- K1: Q/K projection tiles + sim[b,n] -------------- */
/* grid (32, 4), block 128: wave = one 32-row tile x one head.
   Per k-step (32): 4 A-frag + 8 B-frag loads feed 16 WMMAs
   (each weight fragment is reused across both 16-row sub-tiles).       */
__global__ __launch_bounds__(128)
void qk_sim_kernel(const bf16* __restrict__ qbf,  const bf16* __restrict__ kvbf,
                   const bf16* __restrict__ wqbf, const bf16* __restrict__ wkvbf,
                   const float* __restrict__ bq,  const float* __restrict__ bkv,
                   float* __restrict__ sim) {
  const int lane = threadIdx.x & 31;
  const int wave = threadIdx.x >> 5;
  const int n    = blockIdx.y * 4 + wave;   /* head 0..15 */
  const int bt   = blockIdx.x;              /* 32-row batch tile */
  const int l15  = lane & 15;

  const bf16* aq0p = qbf  + (size_t)(bt * 32 + l15)      * IN_DIM;
  const bf16* aq1p = qbf  + (size_t)(bt * 32 + 16 + l15) * IN_DIM;
  const bf16* ak0p = kvbf + (size_t)(bt * 32 + l15)      * IN_DIM;
  const bf16* ak1p = kvbf + (size_t)(bt * 32 + 16 + l15) * IN_DIM;

  v8f accQ[2][4] = {};
  v8f accK[2][4] = {};

  for (int m0 = 0; m0 < IN_DIM; m0 += 32) {
    v16bf aq0 = load_a_frag(aq0p, m0, lane);
    v16bf aq1 = load_a_frag(aq1p, m0, lane);
    v16bf ak0 = load_a_frag(ak0p, m0, lane);
    v16bf ak1 = load_a_frag(ak1p, m0, lane);
#pragma unroll
    for (int t = 0; t < 4; ++t) {
      const bf16* bqp = wqbf  + (size_t)(n * 64  + t * 16 + l15) * IN_DIM;
      const bf16* bkp = wkvbf + (size_t)(n * 128 + t * 16 + l15) * IN_DIM;
      v16bf bq16 = load_b_frag(bqp, m0, lane);
      v16bf bk16 = load_b_frag(bkp, m0, lane);
      accQ[0][t] = __builtin_amdgcn_wmma_f32_16x16x32_bf16(
          false, aq0, false, bq16, (short)0, accQ[0][t], false, false);
      accQ[1][t] = __builtin_amdgcn_wmma_f32_16x16x32_bf16(
          false, aq1, false, bq16, (short)0, accQ[1][t], false, false);
      accK[0][t] = __builtin_amdgcn_wmma_f32_16x16x32_bf16(
          false, ak0, false, bk16, (short)0, accK[0][t], false, false);
      accK[1][t] = __builtin_amdgcn_wmma_f32_16x16x32_bf16(
          false, ak1, false, bk16, (short)0, accK[1][t], false, false);
    }
  }

  /* C/D layout: lane holds col N=l15; VGPR r -> row M = r + (lane<16?0:8).
     sim[b,n] = sum_d (q+bq)(k+bkv): per-lane products, then 16-lane rowsum. */
#pragma unroll
  for (int u = 0; u < 2; ++u) {
    float srow[8] = {0.f, 0.f, 0.f, 0.f, 0.f, 0.f, 0.f, 0.f};
#pragma unroll
    for (int t = 0; t < 4; ++t) {
      float bqv   = bq [n * 64  + t * 16 + l15];
      float bkv_v = bkv[n * 128 + t * 16 + l15];
#pragma unroll
      for (int r = 0; r < 8; ++r)
        srow[r] += (accQ[u][t][r] + bqv) * (accK[u][t][r] + bkv_v);
    }
#pragma unroll
    for (int r = 0; r < 8; ++r) {
      float s = srow[r];
      s += __shfl_xor(s, 1);
      s += __shfl_xor(s, 2);
      s += __shfl_xor(s, 4);
      s += __shfl_xor(s, 8);
      if (l15 == 0) {
        int M = (lane < 16) ? r : (8 + r);
        sim[(size_t)(bt * 32 + u * 16 + M) * NH + n] = s;
      }
    }
  }
}

/* ---------------- K2: Vs[b,n] = kv_input . Wvsum^T + bkvvs --------- */
__global__ __launch_bounds__(32)
void vs_kernel(const bf16* __restrict__ kvbf, const bf16* __restrict__ wvs,
               const float* __restrict__ bkvvs, float* __restrict__ Vs) {
  const int lane = threadIdx.x & 31;
  const int l15  = lane & 15;
  const int bt   = blockIdx.x;
  const int rowg = bt * 16 + l15;
  const bf16* ap = kvbf + (size_t)rowg * IN_DIM;
  const bf16* bp = wvs + (size_t)l15 * IN_DIM;  /* col N = head = l15 */

  v8f acc = {};
  for (int m0 = 0; m0 < IN_DIM; m0 += 32) {
    v16bf a = load_a_frag(ap, m0, lane);
    v16bf b = load_b_frag(bp, m0, lane);
    acc = __builtin_amdgcn_wmma_f32_16x16x32_bf16(
        false, a, false, b, (short)0, acc, false, false);
  }
  float bv = bkvvs[l15];
#pragma unroll
  for (int r = 0; r < 8; ++r) {
    int M = (lane < 16) ? r : (8 + r);
    Vs[(size_t)(bt * 16 + M) * NH + l15] = acc[r] + bv;
  }
}

/* ---------------- epilogue kernels --------------------------------- */
__global__ void softmax_kernel(const float* __restrict__ sim,
                               float* __restrict__ att) {
  int b = blockIdx.x * blockDim.x + threadIdx.x;
  if (b >= B_SZ) return;
  float v[NH];
  float mx = -1e30f;
  for (int n = 0; n < NH; ++n) { v[n] = sim[(size_t)b * NH + n]; mx = fmaxf(mx, v[n]); }
  float s = 0.f;
  for (int n = 0; n < NH; ++n) { v[n] = expf(v[n] - mx); s += v[n]; }
  float inv = 1.0f / s;
  for (int n = 0; n < NH; ++n) att[(size_t)b * NH + n] = v[n] * inv;
}

__global__ void achunk_kernel(const float* __restrict__ att,
                              float* __restrict__ Ach) {
  int tid = threadIdx.x;       /* 64 threads: (t,n) */
  if (tid >= TF * NH) return;
  int t = tid >> 4, n = tid & 15;
  float s = 0.f;
  for (int j = 0; j < CHUNK; ++j) s += att[(size_t)(t * CHUNK + j) * NH + n];
  Ach[t * NH + n] = s;
}

__global__ void pooled_kernel(const float* __restrict__ Vs,
                              const float* __restrict__ Ach,
                              float* __restrict__ x) {
  int i = blockIdx.x * blockDim.x + threadIdx.x;
  if (i >= B_SZ) return;
  float vs[NH];
  for (int n = 0; n < NH; ++n) vs[n] = Vs[(size_t)i * NH + n];
  const float scale = 1.0f / (float)(CHUNK * D);
  for (int t = 0; t < TF; ++t) {
    float p = 0.f;
    for (int n = 0; n < NH; ++n) p += vs[n] * Ach[t * NH + n];
    p *= scale;
    x[(size_t)i * TF + t] = (p >= 0.f) ? p : SLOPE * p;
  }
}

__device__ __forceinline__ float block_sum(float v, float* smem, int tid) {
  v += __shfl_xor(v, 1);  v += __shfl_xor(v, 2);  v += __shfl_xor(v, 4);
  v += __shfl_xor(v, 8);  v += __shfl_xor(v, 16);
  if ((tid & 31) == 0) smem[tid >> 5] = v;
  __syncthreads();
  if (tid < 32) {
    float r = smem[tid];
    r += __shfl_xor(r, 1);  r += __shfl_xor(r, 2);  r += __shfl_xor(r, 4);
    r += __shfl_xor(r, 8);  r += __shfl_xor(r, 16);
    if (tid == 0) smem[0] = r;
  }
  __syncthreads();
  float r = smem[0];
  __syncthreads();
  return r;
}

__global__ __launch_bounds__(1024)
void final_kernel(const float* __restrict__ x,
                  const float* __restrict__ bn_w, const float* __restrict__ bn_b,
                  const float* __restrict__ Wo,   const float* __restrict__ bo,
                  float* __restrict__ out) {
  __shared__ float smem[32];
  int tid = threadIdx.x;   /* one thread per batch row */
  float xv[TF];
  for (int t = 0; t < TF; ++t) xv[t] = x[(size_t)tid * TF + t];
  float mean[TF], var[TF];
  for (int t = 0; t < TF; ++t) {
    float s  = block_sum(xv[t], smem, tid);
    float s2 = block_sum(xv[t] * xv[t], smem, tid);
    mean[t] = s * (1.0f / B_SZ);
    var[t]  = s2 * (1.0f / B_SZ) - mean[t] * mean[t];
  }
  float xn[TF];
  for (int t = 0; t < TF; ++t)
    xn[t] = (xv[t] - mean[t]) * rsqrtf(var[t] + EPS) * bn_w[t] + bn_b[t];
  for (int c = 0; c < NC; ++c) {
    float o = bo[c];
    for (int t = 0; t < TF; ++t) o += xn[t] * Wo[c * TF + t];
    out[(size_t)tid * NC + c] = o;
  }
}

/* ---------------- host side --------------------------------------- */
extern "C" void kernel_launch(void* const* d_in, const int* in_sizes, int n_in,
                              void* d_out, int out_size, void* d_ws, size_t ws_size,
                              hipStream_t stream) {
  (void)in_sizes; (void)n_in; (void)out_size; (void)ws_size;
  const float* q_in  = (const float*)d_in[0];
  const float* kv_in = (const float*)d_in[1];
  const float* Wq    = (const float*)d_in[2];
  const float* bq    = (const float*)d_in[3];
  const float* Wkv   = (const float*)d_in[4];
  const float* bkv   = (const float*)d_in[5];
  const float* bn_w  = (const float*)d_in[6];
  const float* bn_b  = (const float*)d_in[7];
  const float* Wo    = (const float*)d_in[8];
  const float* bo    = (const float*)d_in[9];
  float* out = (float*)d_out;

  char* ws = (char*)d_ws;
  bf16*  qbf   = (bf16*)(ws + OFF_QBF);
  bf16*  kvbf  = (bf16*)(ws + OFF_KVBF);
  bf16*  wqbf  = (bf16*)(ws + OFF_WQBF);
  bf16*  wkvbf = (bf16*)(ws + OFF_WKVBF);
  bf16*  wvs   = (bf16*)(ws + OFF_WVS);
  float* sim   = (float*)(ws + OFF_SIM);
  float* Vs    = (float*)(ws + OFF_VS);
  float* att   = (float*)(ws + OFF_ATT);
  float* Ach   = (float*)(ws + OFF_ACH);
  float* bkvvs = (float*)(ws + OFF_BKVS);
  float* xbuf  = (float*)(ws + OFF_X);

  const int n_q   = B_SZ * IN_DIM;          /* 2M  */
  const int n_wq  = INNER * IN_DIM;         /* 2M  */
  const int n_wkv = 2 * INNER * IN_DIM;     /* 4M  */

  cvt_bf16_kernel<<<(n_q  + 255) / 256, 256, 0, stream>>>(q_in,  qbf,   n_q);
  cvt_bf16_kernel<<<(n_q  + 255) / 256, 256, 0, stream>>>(kv_in, kvbf,  n_q);
  cvt_bf16_kernel<<<(n_wq + 255) / 256, 256, 0, stream>>>(Wq,    wqbf,  n_wq);
  cvt_bf16_kernel<<<(n_wkv + 255) / 256, 256, 0, stream>>>(Wkv,  wkvbf, n_wkv);
  wvsum_kernel<<<(NH * IN_DIM + 255) / 256, 256, 0, stream>>>(Wkv, wvs, bkvvs, bkv);

  dim3 g1(B_SZ / 32, NH / 4);
  qk_sim_kernel<<<g1, 128, 0, stream>>>(qbf, kvbf, wqbf, wkvbf, bq, bkv, sim);
  vs_kernel<<<B_SZ / 16, 32, 0, stream>>>(kvbf, wvs, bkvvs, Vs);

  softmax_kernel<<<B_SZ / 128, 128, 0, stream>>>(sim, att);
  achunk_kernel<<<1, 64, 0, stream>>>(att, Ach);
  pooled_kernel<<<B_SZ / 128, 128, 0, stream>>>(Vs, Ach, xbuf);
  final_kernel<<<1, 1024, 0, stream>>>(xbuf, bn_w, bn_b, Wo, bo, out);
}
